// KW_HybridBranch_24936580120848
// MI455X (gfx1250) — matrix operand order repaired and checked by reading, not compiled, not run on hardware
//
#include <hip/hip_runtime.h>

// ---------------------------------------------------------------------------
// Fused keyword/VQ kernel for MI455X (gfx1250, wave32, WMMA).
//   out[bn, t] = softmax_v( (khat[bn,:] . Ehat[v,:]) / 0.1 ) @ E[v, t]
// Flash-attention-style streaming over the 49408-entry vocab with online
// softmax; all heavy math in v_wmma_f32_16x16x32_bf16 (fp32 accumulate).
// Round 4: kill scratch spills (inline staging indices, 2 prefetches/thread),
// keep async-to-LDS staging + pipelined fragment loads.
// ---------------------------------------------------------------------------

typedef __attribute__((ext_vector_type(16))) __bf16 v16bf;
typedef __attribute__((ext_vector_type(8)))  float  v8f;
typedef __attribute__((__vector_size__(16))) int    v4i;

#if defined(__has_builtin)
#if __has_builtin(__builtin_amdgcn_global_load_async_to_lds_b128)
#define HAVE_ASYNC_LDS 1
#endif
#if __has_builtin(__builtin_amdgcn_s_wait_asynccnt)
#define HAVE_WAIT_ASYNC 1
#endif
#endif

#define BN      1024     // B*N query rows
#define DD      768
#define TDIM    512
#define VV      49408
#define NPAIR   772      // V / 64  (two 32-row chunks per staging step)
#define NSPLIT  16       // vocab splits (gridDim.y)
#define EPSN    1e-8f
#define SOFT_SCALE 14.426950408889634f   // (1/0.1) * log2(e)  -> use exp2

// workspace layout (bytes)
#define OFF_QT   ((size_t)0)                               // Q bf16 [1024][512] (row-major)
#define OFF_EB   (OFF_QT + (size_t)TDIM * BN * 2)          // E  bf16 [V][512]
#define OFF_SC   (OFF_EB + (size_t)VV * TDIM * 2)          // scale f32 [V]
#define OFF_PART ((OFF_SC + (size_t)VV * 4 + 255) & ~(size_t)255)
#define PART_STRIDE_F 8256   // 16*512 acc + 16 m + 16 l (+pad) floats

// ---------------------------------------------------------------------------
// Prep A: kw = audio @ W + b ; row-normalize; store bf16 Q[row][t] (row-major)
// ---------------------------------------------------------------------------
__global__ void kw_proj_norm(const float* __restrict__ audio,
                             const float* __restrict__ Wp,
                             const float* __restrict__ bp,
                             __bf16* __restrict__ qt) {
  __shared__ float arow[DD];
  __shared__ float red[256];
  const int row = blockIdx.x;          // 0..1023
  const int tid = threadIdx.x;
  for (int d = tid; d < DD; d += 256) arow[d] = audio[(size_t)row * DD + d];
  __syncthreads();
  const int t0 = tid, t1 = tid + 256;
  float a0 = bp[t0], a1 = bp[t1];
  for (int d = 0; d < DD; ++d) {
    const float av = arow[d];
    a0 = fmaf(av, Wp[(size_t)d * TDIM + t0], a0);
    a1 = fmaf(av, Wp[(size_t)d * TDIM + t1], a1);
  }
  red[tid] = a0 * a0 + a1 * a1;
  __syncthreads();
  for (int s = 128; s > 0; s >>= 1) {
    if (tid < s) red[tid] += red[tid + s];
    __syncthreads();
  }
  const float inv = 1.0f / fmaxf(sqrtf(red[0]), EPSN);
  qt[(size_t)row * TDIM + t0] = (__bf16)(a0 * inv);
  qt[(size_t)row * TDIM + t1] = (__bf16)(a1 * inv);
}

// ---------------------------------------------------------------------------
// Prep B: bf16 copy of embedding + per-row logit scale = 10*log2(e)/||E_v||
// ---------------------------------------------------------------------------
__global__ void emb_prep(const float* __restrict__ emb,
                         __bf16* __restrict__ eb,
                         float* __restrict__ scale) {
  __shared__ float red[256];
  const int v = blockIdx.x;
  const int tid = threadIdx.x;
  const float e0 = emb[(size_t)v * TDIM + tid];
  const float e1 = emb[(size_t)v * TDIM + tid + 256];
  eb[(size_t)v * TDIM + tid]       = (__bf16)e0;
  eb[(size_t)v * TDIM + tid + 256] = (__bf16)e1;
  red[tid] = e0 * e0 + e1 * e1;
  __syncthreads();
  for (int s = 128; s > 0; s >>= 1) {
    if (tid < s) red[tid] += red[tid + s];
    __syncthreads();
  }
  if (tid == 0) scale[v] = SOFT_SCALE / fmaxf(sqrtf(red[0]), EPSN);
}

// ---------------------------------------------------------------------------
// Main fused kernel. grid = (16 m-groups, 16 vocab splits), 8 waves each.
// Wave w: m-tile = bx*4 + (w&3), chunk-half = w>>2. Each wave keeps a full
// 16x512 fp32 accumulator (32 WMMA C tiles) + its own online-softmax state.
// LDS: two 32x512 bf16 chunks of E, XOR-swizzled in 16B blocks.
// ---------------------------------------------------------------------------
__launch_bounds__(256, 1)
__global__ void kw_flash(const __bf16* __restrict__ qt,
                         const __bf16* __restrict__ eb,
                         const float* __restrict__ scale,
                         float* __restrict__ part) {
  __shared__ uint4 smem[4096];              // 64 KB = 2 x (32 rows x 64 16B-blocks)
  const int tid  = threadIdx.x;
  const int lane = tid & 31;
  const int wv   = tid >> 5;                // 0..7
  const int l15  = lane & 15;
  const int g    = (lane >> 4) & 1;
  const int chalf = wv >> 2;                // which chunk of the staged pair
  const int mtile = blockIdx.x * 4 + (wv & 3);
  const int mbase = mtile * 16;

  // Preload Q B-fragments (one per K=32 slab of t): lane holds column
  // m = mbase + l15, elements t = 32k + 8g + (0..7) and 32k + 16 + 8g + (0..7).
  union UB { v16bf v; uint4 q[2]; };
  UB bq[16];
  const uint4* qt4 = (const uint4*)qt;
#pragma unroll
  for (int k = 0; k < 16; ++k) {
    const int trow = mbase + l15;
    const size_t base = ((size_t)trow * TDIM + 32 * k + 8 * g) >> 3;   // uint4 = 8 halves
    bq[k].q[0] = qt4[base];
    bq[k].q[1] = qt4[base + 2];
  }

  v8f acc[32];
#pragma unroll
  for (int i = 0; i < 32; ++i) acc[i] = (v8f){};
  float rmax = -3.0e38f, lsum = 0.0f;

  const uint4* eb4 = (const uint4*)eb;

  for (int p = blockIdx.y; p < NPAIR; p += NSPLIT) {
    __syncthreads();
    // ---- stage 64 vocab rows (two 32-row chunks) into LDS, swizzled ----
    // unit = tid + u*256 in [0,4096): ch = unit>>11, row = (unit>>6)&31,
    // blk = unit&63; indices recomputed inline (no persistent registers).
    const int vbase_pair = p * 64;
#if defined(HAVE_ASYNC_LDS)
#pragma unroll
    for (int u = 0; u < 16; ++u) {
      const int unit = tid + u * 256;
      const int ch = unit >> 11, rem = unit & 2047;
      const int r = rem >> 6, blk = rem & 63;
      const uint4* gsrc = &eb4[(size_t)(vbase_pair + ch * 32 + r) * 64 + blk];
      __builtin_amdgcn_global_load_async_to_lds_b128(
          (v4i*)gsrc, (v4i*)&smem[ch * 2048 + r * 64 + (blk ^ (r & 7))], 0, 0);
    }
    // warm L2/L0 for the next pair while the async DMA completes:
    // 256 threads x 2 x 128B lines cover the full 64 KB exactly once.
    if (p + NSPLIT < NPAIR) {
      const char* nb = (const char*)(eb4 + (size_t)(p + NSPLIT) * 64 * 64);
      __builtin_prefetch(nb + tid * 128, 0, 1);
      __builtin_prefetch(nb + tid * 128 + 32768, 0, 1);
    }
#if defined(HAVE_WAIT_ASYNC)
    __builtin_amdgcn_s_wait_asynccnt(0);
#else
    asm volatile("s_wait_asynccnt 0x0" ::: "memory");
#endif
#else
#pragma unroll
    for (int u = 0; u < 16; ++u) {
      const int unit = tid + u * 256;
      const int ch = unit >> 11, rem = unit & 2047;
      const int r = rem >> 6, blk = rem & 63;
      const uint4 d = eb4[(size_t)(vbase_pair + ch * 32 + r) * 64 + blk];
      smem[ch * 2048 + r * 64 + (blk ^ (r & 7))] = d;
    }
    if (p + NSPLIT < NPAIR) {
      const char* nb = (const char*)(eb4 + (size_t)(p + NSPLIT) * 64 * 64);
      __builtin_prefetch(nb + tid * 128, 0, 1);
      __builtin_prefetch(nb + tid * 128 + 32768, 0, 1);
    }
#endif
    __syncthreads();

    const int buf   = chalf;
    const int cbase = (p * 2 + chalf) * 32;     // first vocab row of my chunk

    // ---- matmul 1: S^T tiles (c x m), reduce over t = 512, pipelined ----
    union UA { v16bf v; uint4 q[2]; };
    v8f s0 = (v8f){}, s1 = (v8f){};
    UA a0b[2], a1b[2];
    auto lda = [&](int k, UA& A0, UA& A1) {
      const int b0 = 4 * k + g, b1 = b0 + 2;
      const int r0 = l15, r1 = 16 + l15;
      A0.q[0] = smem[buf * 2048 + r0 * 64 + (b0 ^ (r0 & 7))];
      A0.q[1] = smem[buf * 2048 + r0 * 64 + (b1 ^ (r0 & 7))];
      A1.q[0] = smem[buf * 2048 + r1 * 64 + (b0 ^ (r1 & 7))];
      A1.q[1] = smem[buf * 2048 + r1 * 64 + (b1 ^ (r1 & 7))];
    };
    lda(0, a0b[0], a1b[0]);
#pragma unroll
    for (int k = 0; k < 16; ++k) {
      const int cur = k & 1;
      if (k < 15) lda(k + 1, a0b[cur ^ 1], a1b[cur ^ 1]);
      s0 = __builtin_amdgcn_wmma_f32_16x16x32_bf16(false, a0b[cur].v, false,
                                                   bq[k].v, (short)0, s0,
                                                   false, false);
      s1 = __builtin_amdgcn_wmma_f32_16x16x32_bf16(false, a1b[cur].v, false,
                                                   bq[k].v, (short)0, s1,
                                                   false, false);
    }

    // ---- online softmax (lane l15 == row m; group g covers c%16 in [8g,8g+8)) ----
    const float4* scA = (const float4*)(scale + cbase + g * 8);
    const float4* scB = (const float4*)(scale + cbase + 16 + g * 8);
    const float4 sA0 = scA[0], sA1 = scA[1], sB0 = scB[0], sB1 = scB[1];
    const float sc0[8] = {sA0.x, sA0.y, sA0.z, sA0.w, sA1.x, sA1.y, sA1.z, sA1.w};
    const float sc1[8] = {sB0.x, sB0.y, sB0.z, sB0.w, sB1.x, sB1.y, sB1.z, sB1.w};
    float lg0[8], lg1[8];
    float cmax = -3.0e38f;
#pragma unroll
    for (int r = 0; r < 8; ++r) {
      lg0[r] = s0[r] * sc0[r];
      lg1[r] = s1[r] * sc1[r];
      cmax = fmaxf(cmax, fmaxf(lg0[r], lg1[r]));
    }
    cmax = fmaxf(cmax, __shfl_xor(cmax, 16, 32));   // merge the two c half-sets
    const float mnew = fmaxf(rmax, cmax);
    const float f = exp2f(rmax - mnew);             // acc rescale factor (row m)
    float p0[8], p1[8], psum = 0.0f;
#pragma unroll
    for (int r = 0; r < 8; ++r) {
      p0[r] = exp2f(lg0[r] - mnew);
      p1[r] = exp2f(lg1[r] - mnew);
      psum += p0[r] + p1[r];
    }
    lsum = lsum * f + psum;
    rmax = mnew;

    // rescale accumulator: VGPR r holds row (8*g_lane + r); broadcast f(row)
#pragma unroll
    for (int r = 0; r < 8; ++r) {
      const float fr = __shfl(f, r + ((lane & 16) >> 1), 32);
#pragma unroll
      for (int t = 0; t < 32; ++t) acc[t][r] *= fr;
    }

    // P (S^T D-tiles) -> A-fragment, purely lane-local per the WMMA K-map
    v16bf pav;
#pragma unroll
    for (int e = 0; e < 16; ++e)
      pav[e] = (__bf16)((e < 8) ? p0[e] : p1[e - 8]);

    // ---- matmul 2: acc += P @ E  (B rows = vocab, cols = t), pipelined ----
    const __bf16* sbf = (const __bf16*)&smem[buf * 2048];
    UA bb[2];
    auto ldb = [&](int t, UA& B) {
      const int col = 16 * t + l15;
#pragma unroll
      for (int e = 0; e < 16; ++e) {
        const int c = (e < 8) ? (8 * g + e) : (16 + 8 * g + (e - 8));
        B.v[e] = sbf[(c * 64 + ((col >> 3) ^ (c & 7))) * 8 + (col & 7)];
      }
    };
    ldb(0, bb[0]);
#pragma unroll
    for (int t = 0; t < 32; ++t) {
      const int cur = t & 1;
      if (t < 31) ldb(t + 1, bb[cur ^ 1]);
      acc[t] = __builtin_amdgcn_wmma_f32_16x16x32_bf16(false, pav, false,
                                                       bb[cur].v, (short)0,
                                                       acc[t], false, false);
    }
  }

  // ---- write partial: unnormalized acc + per-row (m, l) ----
  const float ltot = lsum + __shfl_xor(lsum, 16, 32);
  const int slot = blockIdx.y * 2 + chalf;           // 0..31
  float* pb = part + (size_t)(mtile * 32 + slot) * PART_STRIDE_F;
#pragma unroll
  for (int t = 0; t < 32; ++t) {
#pragma unroll
    for (int r = 0; r < 8; ++r) {
      const int m = 8 * g + r;
      pb[m * TDIM + 16 * t + l15] = acc[t][r];
    }
  }
  if (lane < 16) {
    pb[8192 + l15] = rmax;
    pb[8208 + l15] = ltot;
  }
}

// ---------------------------------------------------------------------------
// Combine 32 partials per 16-row m-tile: log-sum-exp merge, then normalize.
// ---------------------------------------------------------------------------
__global__ void kw_combine(const float* __restrict__ part,
                           float* __restrict__ out) {
  __shared__ float mm[32][16], ll[32][16], wgt[32][16], rowL[16];
  const int mtile = blockIdx.x;
  const int tid = threadIdx.x;
  const float* pb = part + (size_t)mtile * 32 * PART_STRIDE_F;
  for (int i = tid; i < 512; i += 256) {
    const int s = i >> 4, r = i & 15;
    mm[s][r] = pb[(size_t)s * PART_STRIDE_F + 8192 + r];
    ll[s][r] = pb[(size_t)s * PART_STRIDE_F + 8208 + r];
  }
  __syncthreads();
  if (tid < 16) {
    float M = -3.0e38f;
    for (int s = 0; s < 32; ++s) M = fmaxf(M, mm[s][tid]);
    float L = 0.0f;
    for (int s = 0; s < 32; ++s) {
      const float w = exp2f(mm[s][tid] - M);
      wgt[s][tid] = w;
      L += ll[s][tid] * w;
    }
    rowL[tid] = L;
  }
  __syncthreads();
  for (int idx = tid; idx < 16 * TDIM; idx += 256) {
    const int r = idx >> 9, c = idx & (TDIM - 1);
    float sum = 0.0f;
    for (int s = 0; s < 32; ++s)
      sum += pb[(size_t)s * PART_STRIDE_F + r * TDIM + c] * wgt[s][r];
    out[(size_t)(mtile * 16 + r) * TDIM + c] = sum / rowL[r];
  }
}

// ---------------------------------------------------------------------------
extern "C" void kernel_launch(void* const* d_in, const int* in_sizes, int n_in,
                              void* d_out, int out_size, void* d_ws, size_t ws_size,
                              hipStream_t stream) {
  const float* audio = (const float*)d_in[0];   // [128,8,768]
  const float* Wp    = (const float*)d_in[1];   // [768,512]
  const float* bp    = (const float*)d_in[2];   // [512]
  const float* emb   = (const float*)d_in[3];   // [49408,512]
  float* out = (float*)d_out;                   // [128,8,512] f32

  char* ws = (char*)d_ws;
  __bf16* qt   = (__bf16*)(ws + OFF_QT);
  __bf16* eb   = (__bf16*)(ws + OFF_EB);
  float* scale = (float*)(ws + OFF_SC);
  float* part  = (float*)(ws + OFF_PART);

  kw_proj_norm<<<BN, 256, 0, stream>>>(audio, Wp, bp, qt);
  emb_prep<<<VV, 256, 0, stream>>>(emb, eb, scale);
  kw_flash<<<dim3(16, NSPLIT), 256, 0, stream>>>(qt, eb, scale, part);
  kw_combine<<<64, 256, 0, stream>>>(part, out);
}
